// ExpertChoiceRouter_18451179504168
// MI455X (gfx1250) — matrix operand order repaired and checked by reading, not compiled, not run on hardware
//
#include <hip/hip_runtime.h>

// ---------------------------------------------------------------------------
// ExpertChoiceRouter for MI455X (gfx1250, wave32, WMMA).
//
// Outputs concatenated flat in d_out (return order of the reference):
//   [0      , 32768 )  expert_weights (4,64,128) fp32
//   [32768  , 65536 )  token_indices  (4,64,128) int32 (raw bits)
//   [65536  , 1114112) router_logits  (4,4096,64) fp32
//   [1114112, 1114113) capacity = 128 (int32 raw bits)
//
// Roofline: GEMM1 (16384x2048 @ 2048x128) = 8.6 GFLOP over 134 MB of fp32
// activations -> ~5.8 us HBM floor at 23.3 TB/s; bf16 WMMA (K=32/instr) is
// required to keep the kernel bandwidth-bound. Weights are converted +
// transposed once to bf16 in workspace (512 KB, L2-resident).
// ---------------------------------------------------------------------------

typedef __attribute__((ext_vector_type(16))) __bf16          v16bf;
typedef __attribute__((ext_vector_type(16))) unsigned short  v16u16;
typedef __attribute__((ext_vector_type(8)))  unsigned short  v8u16;
typedef __attribute__((ext_vector_type(8)))  unsigned int    v8u32;
typedef __attribute__((ext_vector_type(8)))  float           v8f;

#define TOK   16384   // B*S
#define DDIM  2048
#define HDIM  128
#define EDIM  64
#define SDIM  4096
#define CAP   128

#if __has_builtin(__builtin_amdgcn_rcpf)
#define FAST_RCP(x) __builtin_amdgcn_rcpf(x)
#else
#define FAST_RCP(x) (1.0f / (x))
#endif

#if __has_builtin(__builtin_amdgcn_sched_barrier)
#define SCHED_BARRIER() __builtin_amdgcn_sched_barrier(0)
#else
#define SCHED_BARRIER()
#endif

__device__ __forceinline__ unsigned short f2bf_u16(float f) {
  // fp32 -> bf16, round-to-nearest-even (scalar path)
  unsigned u = __builtin_bit_cast(unsigned, f);
  unsigned r = u + 0x7FFFu + ((u >> 16) & 1u);
  return (unsigned short)(r >> 16);
}

__device__ __forceinline__ unsigned pk_bf16(float lo, float hi) {
  // Pack two fp32 -> two bf16 (round-half-up) in 3 VALU ops:
  // v_add_u32 x2 (rounding bias) + v_perm_b32 (select the two high halves).
  unsigned ulo = __builtin_bit_cast(unsigned, lo) + 0x8000u;
  unsigned uhi = __builtin_bit_cast(unsigned, hi) + 0x8000u;
#if __has_builtin(__builtin_amdgcn_perm)
  // byte sel: dst[0]=lo.b2 dst[1]=lo.b3 dst[2]=hi.b2 dst[3]=hi.b3
  return __builtin_amdgcn_perm(uhi, ulo, 0x07060302u);
#else
  return (ulo >> 16) | (uhi & 0xFFFF0000u);
#endif
}

// ---------------------------------------------------------------------------
// Kernel 1: convert + transpose router weights once into workspace.
//   w1 (D,H) fp32  -> w1T (H,D) bf16   (512 KB, stays hot in 192 MB L2)
//   w2 (H,E) fp32  -> w2T (E,H) bf16   (16 KB)
// ---------------------------------------------------------------------------
__global__ __launch_bounds__(256) void cvt_weights(
    const float* __restrict__ w1, const float* __restrict__ w2,
    unsigned short* __restrict__ w1T, unsigned short* __restrict__ w2T) {
  int idx = blockIdx.x * 256 + threadIdx.x;
  if (idx < DDIM * HDIM) {
    int d = idx >> 7, h = idx & (HDIM - 1);
    w1T[(size_t)h * DDIM + d] = f2bf_u16(w1[idx]);
  } else {
    int j = idx - DDIM * HDIM;
    if (j < HDIM * EDIM) {
      int h = j >> 6, e = j & (EDIM - 1);
      w2T[(size_t)e * HDIM + h] = f2bf_u16(w2[j]);
    }
  }
}

// ---------------------------------------------------------------------------
// Kernel 2: router_logits = silu(hs @ w1) @ w2, bf16 WMMA, fp32 accumulate.
// 128 tokens per workgroup, 8 waves; wave w owns token rows [w*16, w*16+16).
// Double-buffered LDS stage of the shared w1T K-chunk (one barrier/chunk,
// next chunk's global load overlaps the 8 WMMAs of the current chunk).
// ---------------------------------------------------------------------------
__global__ __launch_bounds__(256) void router_gemm(
    const float* __restrict__ hs, const unsigned short* __restrict__ w1T,
    const unsigned short* __restrict__ w2T, float* __restrict__ logits) {
  __shared__ unsigned short ldsB[2][HDIM * 32];  // staged w1T chunks, 2 x 8 KB
  __shared__ unsigned short ldsH[128 * HDIM];    // silu hidden tile, 32 KB

  const int tid  = threadIdx.x;
  const int w    = tid >> 5;        // wave 0..7
  const int l    = tid & 31;        // lane
  const int lr   = l & 15;          // row (A) / col (B,C) selector
  const int lh   = l >> 4;          // lane half
  const int mbase = blockIdx.x * 128;
  const int khalf = lh * 8;         // unused; A-frag uses koff (contiguous K)
  const int koff  = lh * 16;        // B-frag K sub-offset

  const float* arow = hs + (size_t)(mbase + w * 16 + lr) * DDIM;

  // Per-thread staging slot: col = tid/2, half = tid&1 (256 threads x 32B = 8KB)
  const unsigned short* gstage = w1T + (size_t)(tid >> 1) * DDIM + (tid & 1) * 16;
  const int sslot = (tid >> 1) * 32 + (tid & 1) * 16;

  v8f acc[8];
  #pragma unroll
  for (int t = 0; t < 8; ++t) acc[t] = (v8f){0.f,0.f,0.f,0.f,0.f,0.f,0.f,0.f};

  v16u16 stage = *(const v16u16*)(gstage);  // chunk kc=0
  int buf = 0;

  for (int kc = 0; kc < DDIM; kc += 32) {
    *(v16u16*)(&ldsB[buf][sslot]) = stage;   // publish this chunk
    __syncthreads();

    if (kc + 32 < DDIM)                       // issue next chunk's global load
      stage = *(const v16u16*)(gstage + kc + 32);

    // A fragment: 16 fp32 -> bf16 on the fly (lane-half contiguous K layout).
    const float* ap = arow + kc + koff;
    float4 fa = *(const float4*)(ap);
    float4 fb = *(const float4*)(ap + 4);
    float4 fc = *(const float4*)(ap + 8);
    float4 fd = *(const float4*)(ap + 12);
    v8u32 au;
    au[0] = pk_bf16(fa.x, fa.y); au[1] = pk_bf16(fa.z, fa.w);
    au[2] = pk_bf16(fb.x, fb.y); au[3] = pk_bf16(fb.z, fb.w);
    au[4] = pk_bf16(fc.x, fc.y); au[5] = pk_bf16(fc.z, fc.w);
    au[6] = pk_bf16(fd.x, fd.y); au[7] = pk_bf16(fd.z, fd.w);
    v16bf a = __builtin_bit_cast(v16bf, au);

    // Preload all 8 B fragments; sched_barrier keeps the ds_loads issued
    // ahead of the WMMA chain so the dscnt waits relax and WMMAs drain
    // back-to-back instead of one-load-one-wait-one-wmma.
    v16u16 breg[8];
    #pragma unroll
    for (int t = 0; t < 8; ++t)
      breg[t] = *(const v16u16*)(&ldsB[buf][(t * 16 + lr) * 32 + koff]);
    SCHED_BARRIER();

    #pragma unroll
    for (int t = 0; t < 8; ++t)
      acc[t] = __builtin_amdgcn_wmma_f32_16x16x32_bf16(
          false, a, false, __builtin_bit_cast(v16bf, breg[t]),
          (short)0, acc[t], false, false);

    buf ^= 1;
  }

  // SiLU in registers, spill hidden tile to LDS as bf16 (C-layout -> row-major).
  #pragma unroll
  for (int t = 0; t < 8; ++t) {
    #pragma unroll
    for (int i = 0; i < 8; ++i) {
      float x = acc[t][i];
      float s = x * FAST_RCP(1.0f + __expf(-x));  // silu, fast reciprocal
      int r = w * 16 + lh * 8 + i;                // C layout: M = (l/16)*8 + i
      ldsH[r * HDIM + t * 16 + lr] = f2bf_u16(s); // N = l%16
    }
  }
  __syncthreads();

  // GEMM2: logits(16x64) = h(16x128) @ w2(128x64), all bf16 WMMA.
  v8f d[4];
  #pragma unroll
  for (int t = 0; t < 4; ++t) d[t] = (v8f){0.f,0.f,0.f,0.f,0.f,0.f,0.f,0.f};

  #pragma unroll
  for (int kc = 0; kc < HDIM; kc += 32) {
    int hbase = (w * 16 + lr) * HDIM + kc + koff;
    v8u16 lo = *(const v8u16*)(&ldsH[hbase]);
    v8u16 hi = *(const v8u16*)(&ldsH[hbase + 8]);
    v16u16 am = __builtin_shufflevector(lo, hi, 0,1,2,3,4,5,6,7,8,9,10,11,12,13,14,15);
    v16bf a2 = __builtin_bit_cast(v16bf, am);
    #pragma unroll
    for (int t = 0; t < 4; ++t) {
      int col = t * 16 + lr;
      v16bf b2 = __builtin_bit_cast(
          v16bf, *(const v16u16*)(w2T + (size_t)col * HDIM + kc + koff));
      d[t] = __builtin_amdgcn_wmma_f32_16x16x32_bf16(
          false, a2, false, b2, (short)0, d[t], false, false);
    }
  }

  #pragma unroll
  for (int t = 0; t < 4; ++t) {
    #pragma unroll
    for (int i = 0; i < 8; ++i) {
      int m = mbase + w * 16 + lh * 8 + i;
      logits[(size_t)m * EDIM + t * 16 + lr] = d[t][i];
    }
  }
}

// ---------------------------------------------------------------------------
// Kernel 3: per (b,e) softmax over tokens + top-128 + renormalize.
// One 256-thread WG per expert column; full bitonic sort of 4096 in LDS
// (descending value, ascending index on ties -> matches jax.lax.top_k).
// ---------------------------------------------------------------------------
__global__ __launch_bounds__(256) void router_topk(
    const float* __restrict__ logits, float* __restrict__ w_out,
    int* __restrict__ i_out, int* __restrict__ cap_out) {
  __shared__ float sval[SDIM];
  __shared__ int   sidx[SDIM];
  __shared__ float sred[256];
  __shared__ float sp[CAP];

  const int tid = threadIdx.x;
  const int b   = blockIdx.x >> 6;
  const int e   = blockIdx.x & 63;

  float lmax = -3.402823466e+38f;
  for (int s = tid; s < SDIM; s += 256) {
    float v = logits[((size_t)b * SDIM + s) * EDIM + e];
    sval[s] = v; sidx[s] = s;
    lmax = fmaxf(lmax, v);
  }
  sred[tid] = lmax;
  __syncthreads();
  for (int off = 128; off > 0; off >>= 1) {
    if (tid < off) sred[tid] = fmaxf(sred[tid], sred[tid + off]);
    __syncthreads();
  }
  float gmax = sred[0];
  __syncthreads();

  float lsum = 0.f;
  for (int s = tid; s < SDIM; s += 256) lsum += __expf(sval[s] - gmax);
  sred[tid] = lsum;
  __syncthreads();
  for (int off = 128; off > 0; off >>= 1) {
    if (tid < off) sred[tid] += sred[tid + off];
    __syncthreads();
  }
  float denom = sred[0];
  __syncthreads();

  // Bitonic sort, total order: value desc, index asc.
  for (int k = 2; k <= SDIM; k <<= 1) {
    for (int j = k >> 1; j > 0; j >>= 1) {
      for (int p = tid; p < SDIM / 2; p += 256) {
        int i = ((p / j) * 2 * j) + (p % j);
        int x = i + j;
        bool dir = ((i & k) == 0);                  // descending block
        float vi = sval[i], vx = sval[x];
        int   ii = sidx[i], ix = sidx[x];
        bool before = (vi > vx) || (vi == vx && ii < ix);
        if (dir != before) {
          sval[i] = vx; sval[x] = vi;
          sidx[i] = ix; sidx[x] = ii;
        }
      }
      __syncthreads();
    }
  }

  float p = 0.f;
  if (tid < CAP) { p = __expf(sval[tid] - gmax) / denom; sp[tid] = p; }
  sred[tid] = p;
  __syncthreads();
  for (int off = 128; off > 0; off >>= 1) {
    if (tid < off) sred[tid] += sred[tid + off];
    __syncthreads();
  }
  float s128 = sred[0];
  if (tid < CAP) {
    int o = blockIdx.x * CAP + tid;
    w_out[o] = sp[tid] / (s128 + 1e-9f);
    i_out[o] = sidx[tid];
  }
  if (blockIdx.x == 0 && tid == 0) cap_out[0] = CAP;
}

// ---------------------------------------------------------------------------
extern "C" void kernel_launch(void* const* d_in, const int* in_sizes, int n_in,
                              void* d_out, int out_size, void* d_ws, size_t ws_size,
                              hipStream_t stream) {
  (void)in_sizes; (void)n_in; (void)out_size; (void)ws_size;
  const float* hs = (const float*)d_in[0];  // (4,4096,2048)
  const float* w1 = (const float*)d_in[1];  // (2048,128)
  const float* w2 = (const float*)d_in[2];  // (128,64)

  float* out     = (float*)d_out;
  float* wts     = out;                                   // 32768 floats
  int*   idxs    = (int*)(out + 32768);                   // 32768 int32 (raw bits)
  float* logits  = out + 65536;                           // 1048576 floats
  int*   cap     = (int*)(out + 65536 + (size_t)TOK * EDIM);

  unsigned short* w1T = (unsigned short*)d_ws;                                // 512 KB
  unsigned short* w2T = (unsigned short*)((char*)d_ws + (size_t)DDIM*HDIM*2); // 16 KB

  cvt_weights<<<(DDIM*HDIM + HDIM*EDIM + 255) / 256, 256, 0, stream>>>(w1, w2, w1T, w2T);
  router_gemm<<<TOK / 128, 256, 0, stream>>>(hs, w1T, w2T, logits);
  router_topk<<<4 * EDIM, 256, 0, stream>>>(logits, wts, idxs, cap);
}
